// LinearKAN_5239860101394
// MI455X (gfx1250) — matrix operand order repaired and checked by reading, not compiled, not run on hardware
//
#include <hip/hip_runtime.h>
#include <hip/hip_bf16.h>

typedef __attribute__((ext_vector_type(16))) _Float16 v16h;
typedef __attribute__((ext_vector_type(8)))  _Float16 v8h;
typedef __attribute__((ext_vector_type(8)))  float    v8f;

#define B_SZ   1024
#define O_SZ   256
#define I_SZ   256
#define S_SEG  20                 // S segments -> 21 breakpoints/values per (o,i)
#define CHUNK  8                  // in-features staged per LDS tile
#define OTILE  64                 // o-columns per block (4 x 16-wide WMMA tiles per wave)
#define NSUB   (OTILE / 16)       // 4 sub-tiles per wave
#define OSTRIDE (CHUNK * 32 + 4)  // f16 per o-row in LDS; 130 dwords (== 2 mod 64 banks)

// Block = 256 threads = 8 waves covering 128(b) x 64(o) of the output.
// Wave w owns b-rows [b0+16w, b0+16w+16) and ALL 64 o-columns (4 accumulators),
// so each register-built A fragment (hat weights, depends only on b,i) feeds
// 4 v_wmma_f32_16x16x32_f16 ops. values are staged to LDS once per chunk as
// f16, zero-padded to K=32, shared by all 8 waves.
__global__ __launch_bounds__(256) void kan_wmma_kernel(
    const float* __restrict__ x,       // [B, I]
    const float* __restrict__ bp,      // [O, I, S+1] (tiled identical across O; use o=0 rows)
    const float* __restrict__ values,  // [O, I, S+1]
    float* __restrict__ out)           // [B, O]
{
  __shared__ _Float16 vtile[OTILE * OSTRIDE];   // ~34 KB

  const int tid  = (int)threadIdx.x;
  const int wv   = tid >> 5;       // wave in block: 0..7
  const int lane = tid & 31;
  const int half = lane >> 4;      // 0: lanes 0-15, 1: lanes 16-31
  const int m    = lane & 15;

  const int o_strips = O_SZ / OTILE;               // 4
  const int o0 = (blockIdx.x % o_strips) * OTILE;
  const int b0 = (blockIdx.x / o_strips) * 128 + wv * 16;

  v8f accs[NSUB] = {};

  for (int i0 = 0; i0 < I_SZ; i0 += CHUNK) {
    __syncthreads();   // protect vtile from previous chunk's readers

    // ---- stage values tile: OTILE o x CHUNK i x 32 s (f16, zero-padded) -----
#pragma unroll 4
    for (int j = 0; j < (OTILE * CHUNK * 32) / 256; ++j) {   // 64 slots/thread
      int idx = j * 256 + tid;
      int s   = idx & 31;
      int ii  = (idx >> 5) & (CHUNK - 1);
      int o   = idx >> 8;                                    // 0..63
      int sc  = (s <= S_SEG) ? s : S_SEG;                    // clamp: in-range
      const float* src =
          values + ((size_t)(o0 + o) * I_SZ + (i0 + ii)) * (S_SEG + 1) + sc;
      float v = *src;                                        // unconditional load
      vtile[o * OSTRIDE + ii * 32 + s] =
          (s <= S_SEG) ? (_Float16)v : (_Float16)0.0f;       // branchless pad
    }
    __syncthreads();

    // ---- x chunk for this wave's 16 b-rows: two b128 loads ------------------
    const float* xrow = x + (size_t)(b0 + m) * I_SZ + i0;
    float4 x0 = ((const float4*)xrow)[0];
    float4 x1 = ((const float4*)xrow)[1];
    float xs[CHUNK] = {x0.x, x0.y, x0.z, x0.w, x1.x, x1.y, x1.z, x1.w};

#pragma unroll
    for (int ii = 0; ii < CHUNK; ++ii) {
      const int i = i0 + ii;

      // -------- A fragment: hat weights for row b0+m at in-feature i ---------
      float xv = xs[ii];
      int seg = (int)__builtin_floorf((xv + 1.0f) * (0.5f * (float)S_SEG));
      seg = seg < 0 ? 0 : (seg > S_SEG - 1 ? S_SEG - 1 : seg);
      const float* bprow = bp + (size_t)i * (S_SEG + 1);  // o = 0 row (tiled)
      float blo = bprow[seg];
      float bhi = bprow[seg + 1];
      // v_rcp_f32 instead of the IEEE divide chain (1 TRANS op vs ~10 VALU)
      float t = (xv - blo) * __builtin_amdgcn_rcpf(bhi - blo + 1e-8f);
      bool inr = (xv >= blo) && (xv < bhi);               // reference's where-mask
      _Float16 wlo = (_Float16)(inr ? (1.0f - t) : 0.0f);
      _Float16 whi = (_Float16)(inr ? t : 0.0f);

      // 16-bit A 16x32 layout: lane holds row M=lane&15; element e ->
      // K = (e>=8?16:0) + (lane>=16?8:0) + (e&7)
      v16h a;
#pragma unroll
      for (int e = 0; e < 16; ++e) {
        int k = ((e >= 8) ? 16 : 0) + (half ? 8 : 0) + (e & 7);
        a[e] = (k == seg) ? wlo : ((k == seg + 1) ? whi : (_Float16)0.0f);
      }

      // -------- 4 o-sub-tiles reuse the same A fragment ----------------------
      // 16-bit B 32x16 layout: lane holds col N=lane&15; element e ->
      // K = (e>=8?16:0) + (lane>=16?8:0) + (e&7)
#pragma unroll
      for (int n = 0; n < NSUB; ++n) {
        const v8h* bsrc =
            (const v8h*)&vtile[(n * 16 + m) * OSTRIDE + ii * 32 + half * 8];
        v8h bl = bsrc[0];
        v8h bh = bsrc[2];
        v16h bf;
#pragma unroll
        for (int e = 0; e < 8; ++e) { bf[e] = bl[e]; bf[8 + e] = bh[e]; }

        accs[n] = __builtin_amdgcn_wmma_f32_16x16x32_f16(
            /*neg_a=*/false, a, /*neg_b=*/false, bf,
            /*c_mod=*/(short)0, accs[n], /*reuse_a=*/false, /*reuse_b=*/false);
      }
    }
  }

  // ---- store: f32 C/D 16x16 layout: VGPR r -> M = (half?8:0)+r, N = m -------
#pragma unroll
  for (int n = 0; n < NSUB; ++n) {
#pragma unroll
    for (int r = 0; r < 8; ++r) {
      int row = (half ? 8 : 0) + r;
      out[(size_t)(b0 + row) * O_SZ + (o0 + n * 16 + m)] = accs[n][r];
    }
  }
}

extern "C" void kernel_launch(void* const* d_in, const int* in_sizes, int n_in,
                              void* d_out, int out_size, void* d_ws, size_t ws_size,
                              hipStream_t stream) {
  (void)in_sizes; (void)n_in; (void)out_size; (void)d_ws; (void)ws_size;
  const float* x      = (const float*)d_in[0];   // [B, I]
  const float* bp     = (const float*)d_in[1];   // [O, I, S+1]
  const float* values = (const float*)d_in[2];   // [O, I, S+1]
  float* out = (float*)d_out;                    // [B, O]

  // blocks: (B/128) b-strips x (O/64) o-strips = 8 * 4 = 32 blocks, 256 thr each
  const int blocks = (B_SZ / 128) * (O_SZ / OTILE);
  kan_wmma_kernel<<<blocks, 256, 0, stream>>>(x, bp, values, out);
}
